// Attention_10917806866547
// MI455X (gfx1250) — matrix operand order repaired and checked by reading, not compiled, not run on hardware
//
#include <hip/hip_runtime.h>
#include <stdint.h>

#define B_ 4
#define H_ 16
#define S_ 2048
#define D_ 128
#define KT 64
#define NKT (S_ / KT)

typedef __attribute__((ext_vector_type(16))) _Float16 v16h;
typedef __attribute__((ext_vector_type(8)))  _Float16 v8h;
typedef __attribute__((ext_vector_type(4)))  _Float16 v4h;
typedef __attribute__((ext_vector_type(8)))  float    v8f;
typedef __attribute__((ext_vector_type(4)))  float    v4f;
typedef __attribute__((ext_vector_type(4)))  unsigned int u32x4;
typedef __attribute__((ext_vector_type(8)))  int      i32x8;
typedef __attribute__((ext_vector_type(4)))  int      i32x4;

// 1/sqrt(128) * log2(e): score logits kept in log2 domain -> softmax uses raw v_exp_f32
#define QSCALE2 0.12751744154914894f

// ---------------- Pre-pass: K f32 -> f16 (same layout) ----------------
__global__ __launch_bounds__(256) void cvt_k_kernel(const float* __restrict__ in,
                                                    _Float16* __restrict__ outh) {
    const size_t i = ((size_t)blockIdx.x * 256 + threadIdx.x) * 4;
    v4f f = *(const v4f*)(in + i);
    v4h h;
    #pragma unroll
    for (int x = 0; x < 4; ++x) h[x] = (_Float16)f[x];
    *(v4h*)(outh + i) = h;
}

// ---------------- Pre-pass: V f32 [bh][S][D] -> f16 transposed [bh][D][S] ----------------
__global__ __launch_bounds__(256) void cvt_vt_kernel(const float* __restrict__ v,
                                                     _Float16* __restrict__ vt) {
    __shared__ __align__(16) _Float16 tile[64][64];
    const int s0 = blockIdx.x * 64, d0 = blockIdx.y * 64, bh = blockIdx.z;
    const float* vbh = v + (size_t)bh * S_ * D_;
    _Float16*    vtb = vt + (size_t)bh * D_ * S_;
    const int t = threadIdx.x, r = t >> 2, c4 = t & 3;
    #pragma unroll
    for (int i = 0; i < 4; ++i) {
        v4f f = *(const v4f*)(vbh + (size_t)(s0 + r) * D_ + d0 + c4 * 16 + i * 4);
        #pragma unroll
        for (int x = 0; x < 4; ++x) tile[c4 * 16 + i * 4 + x][r] = (_Float16)f[x];
    }
    __syncthreads();
    v8h a  = *(const v8h*)&tile[r][c4 * 16];
    v8h b2 = *(const v8h*)&tile[r][c4 * 16 + 8];
    *(v8h*)(vtb + (size_t)(d0 + r) * S_ + s0 + c4 * 16)     = a;
    *(v8h*)(vtb + (size_t)(d0 + r) * S_ + s0 + c4 * 16 + 8) = b2;
}

// ---------------- TDM helper: 2D tile DMA global->LDS (D# per ISA 8.3/8.4) ----------------
__device__ __forceinline__ void tdm_load_2d(uint32_t lds_addr, uint64_t gaddr,
                                            uint32_t tile0, uint32_t tile1,
                                            uint32_t stride0) {
    u32x4 g0;
    g0[0] = 1u;                                        // count=1, user descriptor
    g0[1] = lds_addr;                                  // LDS byte address
    g0[2] = (uint32_t)gaddr;                           // global_addr[31:0]
    g0[3] = (uint32_t)(gaddr >> 32) | (2u << 30);      // global_addr[56:32] | type=2
    i32x8 g1;
    g1[0] = (int)(1u << 16);                           // workgroup_mask=0, data_size=2B
    g1[1] = (int)((tile0 & 0xFFFFu) << 16);            // abar=0 | tensor_dim0[15:0]
    g1[2] = (int)((tile0 >> 16) | ((tile1 & 0xFFFFu) << 16)); // td0 hi | td1 lo
    g1[3] = (int)((tile1 >> 16) | (tile0 << 16));      // td1 hi | tile_dim0
    g1[4] = (int)(tile1 & 0xFFFFu);                    // tile_dim1 | tile_dim2=0
    g1[5] = (int)stride0;                              // tensor_dim0_stride[31:0]
    g1[6] = 0;                                         // stride0 hi | stride1 lo
    g1[7] = 0;                                         // stride1 hi
    i32x4 z4 = {0, 0, 0, 0};
    i32x8 z8 = {0, 0, 0, 0, 0, 0, 0, 0};
    __builtin_amdgcn_tensor_load_to_lds(g0, g1, z4, z4, z8, 0);
}

// ---------------- Main flash-attention kernel (TDM double-buffered) ----------------
__global__ __launch_bounds__(256) void fa_fwd_tdm(
    const float* __restrict__ q, const _Float16* __restrict__ kh,
    const _Float16* __restrict__ vth, const int* __restrict__ pad,
    const int* __restrict__ attn, float* __restrict__ out)
{
    __shared__ __align__(32) _Float16 bufK[2][KT * D_];    // [key][d]
    __shared__ __align__(32) _Float16 bufVt[2][D_ * KT];   // [d][key]
    __shared__ __align__(32) _Float16 ldsP[8][16 * KT];    // per-wave P bounce

    const int tid  = threadIdx.x;
    const int wave = tid >> 5;
    const int lane = tid & 31;
    const int l16  = lane & 15;
    const int hsel = lane >> 4;

    const int bh    = blockIdx.y;
    const int b     = bh / H_;
    const int qbase = blockIdx.x * 128 + wave * 16;

    const size_t bhOff = (size_t)bh * S_ * D_;
    const float*    qp  = q + bhOff;
    const _Float16* kbp = kh + bhOff;
    const _Float16* vtp = vth + bhOff;   // [D][S] layout
    float*          op  = out + bhOff;

    // ---- Preload Q tile as 4 A-operands (scaled into log2 domain) ----
    v16h Aq[4];
    {
        const float* qrow = qp + (size_t)(qbase + l16) * D_;
        #pragma unroll
        for (int t = 0; t < 4; ++t) {
            const int base0 = t * 32 + hsel * 16;
            const int base1 = t * 32 + hsel * 16 + 8;
            v4f f0 = *(const v4f*)(qrow + base0);
            v4f f1 = *(const v4f*)(qrow + base0 + 4);
            v4f f2 = *(const v4f*)(qrow + base1);
            v4f f3 = *(const v4f*)(qrow + base1 + 4);
            v16h a;
            #pragma unroll
            for (int i = 0; i < 4; ++i) {
                a[i]      = (_Float16)(f0[i] * QSCALE2);
                a[4 + i]  = (_Float16)(f1[i] * QSCALE2);
                a[8 + i]  = (_Float16)(f2[i] * QSCALE2);
                a[12 + i] = (_Float16)(f3[i] * QSCALE2);
            }
            Aq[t] = a;
        }
    }

    v8f O[8];
    #pragma unroll
    for (int g = 0; g < 8; ++g) O[g] = (v8f){};
    float mrow[8], lrow[8];
    #pragma unroll
    for (int r = 0; r < 8; ++r) { mrow[r] = -1e30f; lrow[r] = 0.0f; }

    const uint32_t ldsKAddr[2]  = { (uint32_t)(uintptr_t)&bufK[0][0],
                                    (uint32_t)(uintptr_t)&bufK[1][0] };
    const uint32_t ldsVtAddr[2] = { (uint32_t)(uintptr_t)&bufVt[0][0],
                                    (uint32_t)(uintptr_t)&bufVt[1][0] };
    const uint64_t kGA  = (uint64_t)(uintptr_t)kbp;
    const uint64_t vtGA = (uint64_t)(uintptr_t)vtp;

    // Prime tile 0 (TDM ignores EXEC; issue from wave 0 only)
    if (wave == 0) {
        tdm_load_2d(ldsKAddr[0],  kGA,  D_, KT, D_);          // 64x128, stride D
        tdm_load_2d(ldsVtAddr[0], vtGA, KT, D_, S_);          // 128x64, stride S
    }

    for (int kt = 0; kt < NKT; ++kt) {
        const int kbase = kt * KT;
        const int cur   = kt & 1;
        __syncthreads();  // all waves done reading the buffer about to be overwritten
        if (wave == 0) {
            if (kt + 1 < NKT) {
                const uint64_t kOff = (uint64_t)(kbase + KT);
                tdm_load_2d(ldsKAddr[cur ^ 1],  kGA + kOff * D_ * 2, D_, KT, D_);
                tdm_load_2d(ldsVtAddr[cur ^ 1], vtGA + kOff * 2,     KT, D_, S_);
                __builtin_amdgcn_s_wait_tensorcnt(2);  // current tile landed
            } else {
                __builtin_amdgcn_s_wait_tensorcnt(0);
            }
        }
        __syncthreads();  // publish current tile

        const _Float16* Kb = bufK[cur];
        const _Float16* Vb = bufVt[cur];

        // ---- Scores: 4 column tiles x 4 K-chunks = 16 WMMAs ----
        v8f c[4];
        #pragma unroll
        for (int j = 0; j < 4; ++j) c[j] = (v8f){};
        #pragma unroll
        for (int t = 0; t < 4; ++t) {
            #pragma unroll
            for (int j = 0; j < 4; ++j) {
                v16h bk = *(const v16h*)&Kb[(j * 16 + l16) * D_ + t * 32 + hsel * 16];
                c[j] = __builtin_amdgcn_wmma_f32_16x16x32_f16(false, Aq[t], false, bk,
                                                              (short)0, c[j], false, false);
            }
        }

        // ---- Mask + online softmax (log2 domain) ----
        const int colb = kbase + l16;
        int pm[4];
        #pragma unroll
        for (int j = 0; j < 4; ++j) pm[j] = pad[b * S_ + colb + j * 16];
        float alpha[8];
        #pragma unroll
        for (int r = 0; r < 8; ++r) {
            const int row = qbase + r + hsel * 8;
            float s[4];
            #pragma unroll
            for (int j = 0; j < 4; ++j) {
                const int am = attn[(size_t)row * S_ + colb + j * 16];
                s[j] = (pm[j] != 0 && am != 0) ? c[j][r] : -1e30f;
            }
            float mx = fmaxf(fmaxf(s[0], s[1]), fmaxf(s[2], s[3]));
            #pragma unroll
            for (int off = 8; off >= 1; off >>= 1)
                mx = fmaxf(mx, __shfl_xor(mx, off, 32));
            const float mn = fmaxf(mrow[r], mx);
            const float al = exp2f(mrow[r] - mn);
            float p[4], sum = 0.0f;
            #pragma unroll
            for (int j = 0; j < 4; ++j) { p[j] = exp2f(s[j] - mn); sum += p[j]; }
            #pragma unroll
            for (int off = 8; off >= 1; off >>= 1)
                sum += __shfl_xor(sum, off, 32);
            lrow[r] = lrow[r] * al + sum;
            mrow[r] = mn;
            alpha[r] = al;
            #pragma unroll
            for (int j = 0; j < 4; ++j)
                ldsP[wave][(r + hsel * 8) * KT + j * 16 + l16] = (_Float16)p[j];
        }

        #pragma unroll
        for (int g = 0; g < 8; ++g) {
            #pragma unroll
            for (int r = 0; r < 8; ++r) O[g][r] *= alpha[r];
        }

        // ---- P as two A-operands (16x64) from per-wave LDS (same-wave DS in order) ----
        v16h Ap[2];
        #pragma unroll
        for (int half = 0; half < 2; ++half) {
            const int arow = l16 * KT + half * 32;
            v8h lo = *(const v8h*)&ldsP[wave][arow + hsel * 16];
            v8h hi = *(const v8h*)&ldsP[wave][arow + hsel * 16 + 8];
            #pragma unroll
            for (int i = 0; i < 8; ++i) { Ap[half][i] = lo[i]; Ap[half][8 + i] = hi[i]; }
        }

        // ---- O += P . V : 8 column groups x 2 key halves = 16 WMMAs ----
        #pragma unroll
        for (int g = 0; g < 8; ++g) {
            v16h bv0 = *(const v16h*)&Vb[(g * 16 + l16) * KT + hsel * 16];
            v16h bv1 = *(const v16h*)&Vb[(g * 16 + l16) * KT + 32 + hsel * 16];
            O[g] = __builtin_amdgcn_wmma_f32_16x16x32_f16(false, Ap[0], false, bv0,
                                                          (short)0, O[g], false, false);
            O[g] = __builtin_amdgcn_wmma_f32_16x16x32_f16(false, Ap[1], false, bv1,
                                                          (short)0, O[g], false, false);
        }
    }

    #pragma unroll
    for (int r = 0; r < 8; ++r) {
        const float inv = 1.0f / lrow[r];
        const size_t rowOff = (size_t)(qbase + r + hsel * 8) * D_;
        #pragma unroll
        for (int g = 0; g < 8; ++g)
            op[rowOff + g * 16 + l16] = O[g][r] * inv;
    }
}

// ---------------- Fallback (round-1 kernel, in-LDS conversion, no workspace) ----------------
__global__ __launch_bounds__(256) void fa_fwd_fallback(
    const float* __restrict__ q, const float* __restrict__ k,
    const float* __restrict__ v, const int* __restrict__ pad,
    const int* __restrict__ attn, float* __restrict__ out)
{
    __shared__ __align__(32) _Float16 ldsK[32 * 128];
    __shared__ __align__(32) _Float16 ldsVt[128 * 32];
    __shared__ __align__(32) _Float16 ldsP[8][16 * 32];

    const int tid  = threadIdx.x;
    const int wave = tid >> 5;
    const int lane = tid & 31;
    const int l16  = lane & 15;
    const int hsel = lane >> 4;

    const int bh    = blockIdx.y;
    const int b     = bh / H_;
    const int qbase = blockIdx.x * 128 + wave * 16;

    const size_t bhOff = (size_t)bh * S_ * D_;
    const float* qp = q + bhOff;
    const float* kp = k + bhOff;
    const float* vp = v + bhOff;
    float*       op = out + bhOff;

    v16h Aq[4];
    {
        const float* qrow = qp + (size_t)(qbase + l16) * D_;
        #pragma unroll
        for (int t = 0; t < 4; ++t) {
            const int base0 = t * 32 + hsel * 16;
            const int base1 = t * 32 + hsel * 16 + 8;
            v4f f0 = *(const v4f*)(qrow + base0);
            v4f f1 = *(const v4f*)(qrow + base0 + 4);
            v4f f2 = *(const v4f*)(qrow + base1);
            v4f f3 = *(const v4f*)(qrow + base1 + 4);
            v16h a;
            #pragma unroll
            for (int i = 0; i < 4; ++i) {
                a[i]      = (_Float16)(f0[i] * QSCALE2);
                a[4 + i]  = (_Float16)(f1[i] * QSCALE2);
                a[8 + i]  = (_Float16)(f2[i] * QSCALE2);
                a[12 + i] = (_Float16)(f3[i] * QSCALE2);
            }
            Aq[t] = a;
        }
    }

    v8f O[8];
    #pragma unroll
    for (int g = 0; g < 8; ++g) O[g] = (v8f){};
    float mrow[8], lrow[8];
    #pragma unroll
    for (int r = 0; r < 8; ++r) { mrow[r] = -1e30f; lrow[r] = 0.0f; }

    for (int kt = 0; kt < S_ / 32; ++kt) {
        const int kbase = kt * 32;
        __syncthreads();
        #pragma unroll
        for (int i = 0; i < 4; ++i) {
            const int idx = (tid + i * 256) * 4;
            v4f kf = *(const v4f*)(kp + (size_t)kbase * D_ + idx);
            v4h khx;
            #pragma unroll
            for (int x = 0; x < 4; ++x) khx[x] = (_Float16)kf[x];
            *(v4h*)&ldsK[idx] = khx;
        }
        {
            const int kk = tid >> 3;
            const int cc = tid & 7;
            #pragma unroll
            for (int j = 0; j < 4; ++j) {
                const int n0 = (cc + j * 8) * 4;
                v4f vf = *(const v4f*)(vp + (size_t)(kbase + kk) * D_ + n0);
                #pragma unroll
                for (int x = 0; x < 4; ++x)
                    ldsVt[(n0 + x) * 32 + kk] = (_Float16)vf[x];
            }
        }
        __syncthreads();

        v8f c0 = {}, c1 = {};
        #pragma unroll
        for (int t = 0; t < 4; ++t) {
            v16h b0 = *(const v16h*)&ldsK[l16 * 128 + t * 32 + hsel * 16];
            v16h b1 = *(const v16h*)&ldsK[(16 + l16) * 128 + t * 32 + hsel * 16];
            c0 = __builtin_amdgcn_wmma_f32_16x16x32_f16(false, Aq[t], false, b0,
                                                        (short)0, c0, false, false);
            c1 = __builtin_amdgcn_wmma_f32_16x16x32_f16(false, Aq[t], false, b1,
                                                        (short)0, c1, false, false);
        }

        const int col0 = kbase + l16;
        const int col1 = col0 + 16;
        const int pm0 = pad[b * S_ + col0];
        const int pm1 = pad[b * S_ + col1];
        float alpha[8];
        #pragma unroll
        for (int r = 0; r < 8; ++r) {
            const int row = qbase + r + hsel * 8;
            const int am0 = attn[(size_t)row * S_ + col0];
            const int am1 = attn[(size_t)row * S_ + col1];
            float s0 = (pm0 != 0 && am0 != 0) ? c0[r] : -1e30f;
            float s1 = (pm1 != 0 && am1 != 0) ? c1[r] : -1e30f;
            float mx = fmaxf(s0, s1);
            #pragma unroll
            for (int off = 8; off >= 1; off >>= 1)
                mx = fmaxf(mx, __shfl_xor(mx, off, 32));
            const float mn = fmaxf(mrow[r], mx);
            const float al = exp2f(mrow[r] - mn);
            const float p0 = exp2f(s0 - mn);
            const float p1 = exp2f(s1 - mn);
            float sum = p0 + p1;
            #pragma unroll
            for (int off = 8; off >= 1; off >>= 1)
                sum += __shfl_xor(sum, off, 32);
            lrow[r] = lrow[r] * al + sum;
            mrow[r] = mn;
            alpha[r] = al;
            ldsP[wave][(r + hsel * 8) * 32 + l16]      = (_Float16)p0;
            ldsP[wave][(r + hsel * 8) * 32 + 16 + l16] = (_Float16)p1;
        }

        #pragma unroll
        for (int g = 0; g < 8; ++g) {
            #pragma unroll
            for (int r = 0; r < 8; ++r) O[g][r] *= alpha[r];
        }

        v16h Ap;
        {
            const int arow = l16 * 32;
            v8h lo = *(const v8h*)&ldsP[wave][arow + hsel * 16];
            v8h hi = *(const v8h*)&ldsP[wave][arow + hsel * 16 + 8];
            #pragma unroll
            for (int i = 0; i < 8; ++i) { Ap[i] = lo[i]; Ap[8 + i] = hi[i]; }
        }

        #pragma unroll
        for (int g = 0; g < 8; ++g) {
            v16h bv = *(const v16h*)&ldsVt[(g * 16 + l16) * 32 + hsel * 16];
            O[g] = __builtin_amdgcn_wmma_f32_16x16x32_f16(false, Ap, false, bv,
                                                          (short)0, O[g], false, false);
        }
    }

    #pragma unroll
    for (int r = 0; r < 8; ++r) {
        const float inv = 1.0f / lrow[r];
        const size_t rowOff = (size_t)(qbase + r + hsel * 8) * D_;
        #pragma unroll
        for (int g = 0; g < 8; ++g)
            op[rowOff + g * 16 + l16] = O[g][r] * inv;
    }
}

extern "C" void kernel_launch(void* const* d_in, const int* in_sizes, int n_in,
                              void* d_out, int out_size, void* d_ws, size_t ws_size,
                              hipStream_t stream) {
    const float* q    = (const float*)d_in[0];
    const float* k    = (const float*)d_in[1];
    const float* v    = (const float*)d_in[2];
    const int*   pad  = (const int*)d_in[3];
    const int*   attn = (const int*)d_in[4];
    float*       out  = (float*)d_out;

    const size_t nElem = (size_t)B_ * H_ * S_ * D_;          // 16.78M
    const size_t needH = nElem * sizeof(_Float16);           // 32 MB per tensor

    if (ws_size >= 2 * needH) {
        _Float16* khw  = (_Float16*)d_ws;
        _Float16* vthw = (_Float16*)((char*)d_ws + needH);
        cvt_k_kernel<<<(int)(nElem / 1024), 256, 0, stream>>>(k, khw);
        cvt_vt_kernel<<<dim3(S_ / 64, D_ / 64, B_ * H_), 256, 0, stream>>>(v, vthw);
        fa_fwd_tdm<<<dim3(S_ / 128, B_ * H_), 256, 0, stream>>>(q, khw, vthw, pad, attn, out);
    } else {
        fa_fwd_fallback<<<dim3(S_ / 128, B_ * H_), 256, 0, stream>>>(q, k, v, pad, attn, out);
    }
    (void)in_sizes; (void)n_in; (void)out_size;
}
